// Head_10831907521163
// MI455X (gfx1250) — compile-verified, matrix operand order checked by reading, not run on hardware
//
#include <hip/hip_runtime.h>
#include <hip/hip_bf16.h>

// ---------------------------------------------------------------------------
// Masked single-head attention, MI455X (gfx1250), wave32 + WMMA bf16.
//   B=8, T=2048, E=1024, H=2048.  Compute-bound (~480 GFLOP vs ~15us of HBM),
//   so all three GEMM stages run on v_wmma_f32_16x16x32_bf16.
//   Wave tile 64x32 (4x2 accums): 8 WMMAs per 6 fragment loads.
//   Block 256 thr = 8 waves (2M x 4N) -> 128x128 block tile.
//   K loop: ping-pong double buffering unrolled by 2 -> fixed buffer roles,
//   zero rotation movs; each buffer refill is separated from its consuming
//   WMMA group by the other buffer's 8 WMMAs (covers the 4-slot WMMA->VALU
//   WAR hazard, ISA 7.12.1, with useful work and hides load latency).
// Workspace layout (needs 256 MiB):
//   [0..64M)    Q   bf16 [B,T,H]
//   [64..128M)  K   bf16 [B,T,H]
//   [128..192M) V^T bf16 [B,H,T]   (transposed so P*V loads B-operand K-major)
//   [192..256M) S/P bf16 [B,T,T]   (scores, overwritten in place by softmax)
// ---------------------------------------------------------------------------

typedef __attribute__((ext_vector_type(16))) __bf16 v16bf;
typedef __attribute__((ext_vector_type(8)))  __bf16 v8bf;
typedef __attribute__((ext_vector_type(8)))  float  v8f;
typedef __attribute__((ext_vector_type(8)))  float  v8fl;

#define BB 8
#define TT 2048
#define EE 1024
#define HH 2048

// --- WMMA wrapper ----------------------------------------------------------
__device__ __forceinline__ v8f wmma_bf16(v16bf a, v16bf b, v8f c) {
  // D = A(16x32 bf16) * B(32x16 bf16) + C(16x16 f32)
  return __builtin_amdgcn_wmma_f32_16x16x32_bf16(
      /*neg_a=*/false, a, /*neg_b=*/false, b,
      /*c_mod=*/(short)0, c, /*reuse_a=*/false, /*reuse_b=*/false);
}

// --- A/B fragment loaders (ISA 7.12.2 16-bit A layout; B loaded as B^T) ----
// Lane L<16  : row M=L,       elems 0..7 = K+0..7,  elems 8..15 = K+16..23
// Lane L>=16 : row M=L-16,    elems 0..7 = K+8..15, elems 8..15 = K+24..31
struct LoaderF32 {
  const float* base;
  int ld;
  __device__ __forceinline__ v16bf operator()(int row0, int k0) const {
    const int lane = threadIdx.x & 31;
    const float* p = base + (long)(row0 + (lane & 15)) * ld + k0 + ((lane & 16) ? 8 : 0);
    v8fl x0 = *(const v8fl*)(p);       // K+0..7   (32B aligned)
    v8fl x1 = *(const v8fl*)(p + 16);  // K+16..23
    v16bf f;
#pragma unroll
    for (int i = 0; i < 8; ++i) {
      f[i]     = (__bf16)x0[i];
      f[i + 8] = (__bf16)x1[i];
    }
    return f;
  }
};

struct LoaderBF16 {
  const __bf16* base;
  int ld;
  __device__ __forceinline__ v16bf operator()(int row0, int k0) const {
    const int lane = threadIdx.x & 31;
    const __bf16* p = base + (long)(row0 + (lane & 15)) * ld + k0 + ((lane & 16) ? 8 : 0);
    v8bf lo = *(const v8bf*)(p);        // 16B vector loads
    v8bf hi = *(const v8bf*)(p + 16);
    return __builtin_shufflevector(lo, hi, 0, 1, 2, 3, 4, 5, 6, 7,
                                           8, 9, 10, 11, 12, 13, 14, 15);
  }
};

// issue the 8 WMMAs of one 64x32 wave tile K-step
__device__ __forceinline__ void mma_step(v8f acc[4][2], const v16bf af[4],
                                         v16bf bf0, v16bf bf1) {
#pragma unroll
  for (int i = 0; i < 4; ++i) {
    acc[i][0] = wmma_bf16(af[i], bf0, acc[i][0]);
    acc[i][1] = wmma_bf16(af[i], bf1, acc[i][1]);
  }
}

// Ping-pong pipelined 64x32-per-wave GEMM over K = [0, kend).
// Requires kend to be a multiple of 64 and >= 64.
template <typename LA, typename LB>
__device__ __forceinline__ void gemm_pipelined(v8f acc[4][2], const LA& la,
                                               const LB& lb, int m0, int n0,
                                               int kend) {
#pragma unroll
  for (int i = 0; i < 4; ++i) { acc[i][0] = (v8f){}; acc[i][1] = (v8f){}; }

  v16bf aA[4], b0A, b1A;  // buffer A: even K steps
  v16bf aB[4], b0B, b1B;  // buffer B: odd  K steps
  b0A = lb(n0, 0);  b1A = lb(n0 + 16, 0);
#pragma unroll
  for (int i = 0; i < 4; ++i) aA[i] = la(m0 + i * 16, 0);
  b0B = lb(n0, 32); b1B = lb(n0 + 16, 32);
#pragma unroll
  for (int i = 0; i < 4; ++i) aB[i] = la(m0 + i * 16, 32);

#pragma unroll 1
  for (int k = 64; k < kend; k += 64) {
    __builtin_prefetch(la.base + (long)m0 * la.ld + k + 128, 0, 1);
    __builtin_prefetch(lb.base + (long)n0 * lb.ld + k + 128, 0, 1);
    mma_step(acc, aA, b0A, b1A);             // consume A (k-64)
    b0A = lb(n0, k);  b1A = lb(n0 + 16, k);  // refill A (k)
#pragma unroll
    for (int i = 0; i < 4; ++i) aA[i] = la(m0 + i * 16, k);
    mma_step(acc, aB, b0B, b1B);                       // consume B (k-32)
    b0B = lb(n0, k + 32); b1B = lb(n0 + 16, k + 32);   // refill B (k+32)
#pragma unroll
    for (int i = 0; i < 4; ++i) aB[i] = la(m0 + i * 16, k + 32);
  }
  mma_step(acc, aA, b0A, b1A);
  mma_step(acc, aB, b0B, b1B);
}

// ---------------------------------------------------------------------------
// 1) Projection GEMM: Out[b,t,h] = sum_e X[b,t,e] * W[h,e]   (bf16 out)
//    TRANSPOSE=1 stores Out[b,h,t] instead (used for V).  Compile-time flag
//    so each store path is straight-line (no per-element scalar branching).
// ---------------------------------------------------------------------------
template <int TRANSPOSE>
__global__ void __launch_bounds__(256)
proj_kernel(const float* __restrict__ X, const float* __restrict__ W,
            __bf16* __restrict__ Out) {
  const int b  = blockIdx.z;
  const int w  = threadIdx.x >> 5;
  const int m0 = blockIdx.y * 128 + (w & 1) * 64;   // t
  const int n0 = blockIdx.x * 128 + (w >> 1) * 32;  // h

  v8f acc[4][2];
  LoaderF32 la{X + (long)b * TT * EE, EE};
  LoaderF32 lb{W, EE};
  gemm_pipelined(acc, la, lb, m0, n0, EE);

  // C/D layout: lane<16 -> N=lane, M=VGPRidx; lane>=16 -> N=lane-16, M=VGPRidx+8
  const int lane = threadIdx.x & 31;
  const int nn   = lane & 15;
  const int mb   = (lane & 16) ? 8 : 0;
  __bf16* Ob = Out + (long)b * TT * HH;
#pragma unroll
  for (int i = 0; i < 4; ++i) {
#pragma unroll
    for (int j = 0; j < 2; ++j) {
      const int mi = m0 + i * 16 + mb;
      const int nj = n0 + j * 16 + nn;
      const v8f c = acc[i][j];
#pragma unroll
      for (int r = 0; r < 8; ++r) {
        if (TRANSPOSE) Ob[(long)nj * TT + (mi + r)] = (__bf16)c[r];
        else           Ob[(long)(mi + r) * HH + nj] = (__bf16)c[r];
      }
    }
  }
}

// ---------------------------------------------------------------------------
// 2) Scores: S[b,q,j] = (1/sqrt(E)) * sum_h Q[b,q,h]*K[b,j,h]   (bf16 out)
//    Fully-masked 128x128 tiles (whole tile has j > q+1) are skipped.
// ---------------------------------------------------------------------------
__global__ void __launch_bounds__(256)
scores_kernel(const __bf16* __restrict__ Q, const __bf16* __restrict__ K,
              __bf16* __restrict__ S) {
  if (blockIdx.x > blockIdx.y + 1) return;  // tile fully masked; softmax zeroes it

  const int b  = blockIdx.z;
  const int w  = threadIdx.x >> 5;
  const int m0 = blockIdx.y * 128 + (w & 1) * 64;   // q
  const int n0 = blockIdx.x * 128 + (w >> 1) * 32;  // j

  v8f acc[4][2];
  LoaderBF16 la{Q + (long)b * TT * HH, HH};
  LoaderBF16 lb{K + (long)b * TT * HH, HH};
  gemm_pipelined(acc, la, lb, m0, n0, HH);

  const float scale = 0.03125f;  // 1/sqrt(1024)
  const int lane = threadIdx.x & 31;
  const int nn   = lane & 15;
  const int mb   = (lane & 16) ? 8 : 0;
  __bf16* Sb = S + (long)b * TT * TT;
#pragma unroll
  for (int i = 0; i < 4; ++i) {
#pragma unroll
    for (int j = 0; j < 2; ++j) {
      const int mi = m0 + i * 16 + mb;
      const int nj = n0 + j * 16 + nn;
      const v8f c = acc[i][j];
#pragma unroll
      for (int r = 0; r < 8; ++r)
        Sb[(long)(mi + r) * TT + nj] = (__bf16)(c[r] * scale);
    }
  }
}

// ---------------------------------------------------------------------------
// 3) Masked row softmax, in place (S -> P). One 256-thread block per (b,q).
//    allowed: j <= q+1 (faithful to tril(.., diagonal=1)); others -> P = 0.
// ---------------------------------------------------------------------------
__global__ void __launch_bounds__(256)
softmax_kernel(__bf16* __restrict__ S) {
  const int b = blockIdx.y;
  const int q = blockIdx.x;
  __bf16* row = S + ((long)b * TT + q) * TT;
  const int tid = threadIdx.x;
  const int lim = q + 1;

  float vals[8];
  float m = -__builtin_inff();
#pragma unroll
  for (int i = 0; i < 8; ++i) {
    const int j = tid + i * 256;
    const float v = (j <= lim) ? (float)row[j] : -__builtin_inff();
    vals[i] = v;
    m = fmaxf(m, v);
  }
  __shared__ float red[256];
  red[tid] = m;
  __syncthreads();
  for (int s = 128; s > 0; s >>= 1) {
    if (tid < s) red[tid] = fmaxf(red[tid], red[tid + s]);
    __syncthreads();
  }
  const float rmax = red[0];
  __syncthreads();

  float ex[8];
  float sum = 0.f;
#pragma unroll
  for (int i = 0; i < 8; ++i) {
    ex[i] = (vals[i] == -__builtin_inff()) ? 0.f : __expf(vals[i] - rmax);
    sum += ex[i];
  }
  red[tid] = sum;
  __syncthreads();
  for (int s = 128; s > 0; s >>= 1) {
    if (tid < s) red[tid] += red[tid + s];
    __syncthreads();
  }
  const float inv = 1.0f / red[0];
#pragma unroll
  for (int i = 0; i < 8; ++i)
    row[tid + i * 256] = (__bf16)(ex[i] * inv);
}

// ---------------------------------------------------------------------------
// 4) Output GEMM: O[b,q,h] = sum_j P[b,q,j] * V[b,j,h]  (f32 out)
//    B-operand rows come from V^T[b,h,j]. K-loop trimmed to causal extent,
//    rounded up to 64 so the ping-pong pipeline needs no tail (the extra
//    columns are exactly the masked entries where softmax wrote P == 0).
// ---------------------------------------------------------------------------
__global__ void __launch_bounds__(256)
out_kernel(const __bf16* __restrict__ P, const __bf16* __restrict__ VT,
           float* __restrict__ O) {
  const int b   = blockIdx.z;
  const int m0b = blockIdx.y * 128;
  const int w   = threadIdx.x >> 5;
  const int m0  = m0b + (w & 1) * 64;                 // q
  const int n0  = blockIdx.x * 128 + (w >> 1) * 32;   // h

  int kend = (m0b + 129 + 63) & ~63;  // multiple of 64, >= 192
  if (kend > TT) kend = TT;

  v8f acc[4][2];
  LoaderBF16 la{P  + (long)b * TT * TT, TT};
  LoaderBF16 lb{VT + (long)b * HH * TT, TT};
  gemm_pipelined(acc, la, lb, m0, n0, kend);

  const int lane = threadIdx.x & 31;
  const int nn   = lane & 15;
  const int mb   = (lane & 16) ? 8 : 0;
  float* Ob = O + (long)b * TT * HH;
#pragma unroll
  for (int i = 0; i < 4; ++i) {
#pragma unroll
    for (int j = 0; j < 2; ++j) {
      const int mi = m0 + i * 16 + mb;
      const int nj = n0 + j * 16 + nn;
      const v8f c = acc[i][j];
#pragma unroll
      for (int r = 0; r < 8; ++r)
        Ob[(long)(mi + r) * HH + nj] = c[r];
    }
  }
}

// ---------------------------------------------------------------------------
extern "C" void kernel_launch(void* const* d_in, const int* in_sizes, int n_in,
                              void* d_out, int out_size, void* d_ws, size_t ws_size,
                              hipStream_t stream) {
  const float* key   = (const float*)d_in[0];
  const float* query = (const float*)d_in[1];
  const float* value = (const float*)d_in[2];
  const float* Wk    = (const float*)d_in[3];
  const float* Wq    = (const float*)d_in[4];
  const float* Wv    = (const float*)d_in[5];
  float* out = (float*)d_out;

  const size_t SEG = (size_t)BB * TT * HH * sizeof(__bf16);  // 64 MiB
  char* ws = (char*)d_ws;
  __bf16* Qb = (__bf16*)(ws);
  __bf16* Kb = (__bf16*)(ws + SEG);
  __bf16* VT = (__bf16*)(ws + 2 * SEG);
  __bf16* S  = (__bf16*)(ws + 3 * SEG);  // scores, then probs (in place)

  dim3 blk(256);
  dim3 gproj(HH / 128, TT / 128, BB);
  proj_kernel<0><<<gproj, blk, 0, stream>>>(query, Wq, Qb);
  proj_kernel<0><<<gproj, blk, 0, stream>>>(key,   Wk, Kb);
  proj_kernel<1><<<gproj, blk, 0, stream>>>(value, Wv, VT);

  dim3 gsc(TT / 128, TT / 128, BB);
  scores_kernel<<<gsc, blk, 0, stream>>>(Qb, Kb, S);

  softmax_kernel<<<dim3(TT, BB), blk, 0, stream>>>(S);

  dim3 go(HH / 128, TT / 128, BB);
  out_kernel<<<go, blk, 0, stream>>>(S, VT, out);
}
